// SupplyChainGNN_44401371906104
// MI455X (gfx1250) — matrix operand order repaired and checked by reading, not compiled, not run on hardware
//
#include <hip/hip_runtime.h>
#include <cstdint>

typedef __attribute__((ext_vector_type(16))) _Float16 v16h;
typedef __attribute__((ext_vector_type(8)))  float    v8f;

#define N_NODES 50000
#define F_IN    128
#define HID     64
#define HEADS1  4
#define NGRAPH  512
#define NEDGE   800000

// ---------------------------------------------------------------------------
// WMMA GEMM: C[M,NN] = A[M,K] @ B[K,NN], f32 in/out, f16 multiply, f32 accum.
// One wave per 16x16 output tile. K, NN compile-time so the K loop unrolls.
// ---------------------------------------------------------------------------
template <int K, int NN>
__global__ void gemm_wmma_f16(const float* __restrict__ A,
                              const float* __restrict__ B,
                              float* __restrict__ C) {
  const int lane = threadIdx.x & 31;
  const int l15  = lane & 15;
  const int half = lane >> 4;            // 0: lanes 0-15, 1: lanes 16-31
  const int m0   = blockIdx.x * 16;
  const int n0   = blockIdx.y * 16;

  v8f acc = {};
#pragma unroll
  for (int k0 = 0; k0 < K; k0 += 32) {
    // A fragment (16x32 f16): row = m0+l15; K = k0 + half*16 + {0..15}
    const float* arow = A + (size_t)(m0 + l15) * K + k0 + half * 16;
    v16h a;
#pragma unroll
    for (int i = 0; i < 8; ++i) {
      a[i]     = (_Float16)arow[i];
      a[8 + i] = (_Float16)arow[8 + i];
    }
    // B fragment (32x16 f16): col = n0+l15; rows k0 + half*16 + {0..15}
    const float* bcol = B + (size_t)(k0 + half * 16) * NN + n0 + l15;
    v16h b;
#pragma unroll
    for (int i = 0; i < 16; ++i) {
      b[i] = (_Float16)bcol[(size_t)i * NN];
    }
    acc = __builtin_amdgcn_wmma_f32_16x16x32_f16(
        /*neg_a=*/false, a, /*neg_b=*/false, b,
        /*c_mod=*/(short)0, acc, /*reuse_a=*/false, /*reuse_b=*/false);
  }
  // C/D layout: VGPR r -> row m0 + r + half*8, col n0 + l15
#pragma unroll
  for (int r = 0; r < 8; ++r) {
    C[(size_t)(m0 + r + half * 8) * NN + n0 + l15] = acc[r];
  }
}

// ---------------------------------------------------------------------------
// Per-node attention scalars: asrc[n,h] = <h[n,h,:], a_src[h,:]>, same for dst.
// ---------------------------------------------------------------------------
__global__ void node_att_kernel(const float* __restrict__ h,
                                const float* __restrict__ a_src,
                                const float* __restrict__ a_dst,
                                float* __restrict__ asrc,
                                float* __restrict__ adst,
                                int n, int H, int D) {
  int i = blockIdx.x * blockDim.x + threadIdx.x;    // i = node*H + head
  if (i >= n * H) return;
  int head = i % H;
  const float* hp = h + (size_t)i * D;
  const float* as = a_src + (size_t)head * D;
  const float* ad = a_dst + (size_t)head * D;
  float s = 0.f, d = 0.f;
  for (int j = 0; j < D; ++j) { s += hp[j] * as[j]; d += hp[j] * ad[j]; }
  asrc[i] = s;
  adst[i] = d;
}

// ---------------------------------------------------------------------------
// Order-preserving float<->uint for atomicMax on floats (incl. negatives).
// ---------------------------------------------------------------------------
__device__ __forceinline__ unsigned f2ord(float f) {
  unsigned u = __float_as_uint(f);
  return (u & 0x80000000u) ? ~u : (u | 0x80000000u);
}
__device__ __forceinline__ float ord2f(unsigned k) {
  unsigned u = (k & 0x80000000u) ? (k ^ 0x80000000u) : ~k;
  return __uint_as_float(u);
}

__global__ void fill_u32(unsigned* __restrict__ p, unsigned v, size_t n) {
  size_t i = (size_t)blockIdx.x * blockDim.x + threadIdx.x;
  if (i < n) p[i] = v;
}
__global__ void fill_f32(float* __restrict__ p, float v, size_t n) {
  size_t i = (size_t)blockIdx.x * blockDim.x + threadIdx.x;
  if (i < n) p[i] = v;
}

// ---------------------------------------------------------------------------
// Pass 1: e = leaky_relu(asrc[src]+adst[dst]); stash e; segment max over dst.
// Edges 0..E-1 come from edge_index; edges E..E+n-1 are self-loops.
// ---------------------------------------------------------------------------
__global__ void edge_logits_max(const int* __restrict__ eidx, int Eo, int n,
                                const float* __restrict__ asrc,
                                const float* __restrict__ adst, int H,
                                float* __restrict__ ebuf,
                                unsigned* __restrict__ mmax) {
  int t = blockIdx.x * blockDim.x + threadIdx.x;
  int total = (Eo + n) * H;
  if (t >= total) return;
  int e = t / H, hd = t - e * H;
  int s = (e < Eo) ? eidx[e]       : (e - Eo);
  int d = (e < Eo) ? eidx[Eo + e]  : (e - Eo);
  float v = asrc[(size_t)s * H + hd] + adst[(size_t)d * H + hd];
  v = (v > 0.f) ? v : 0.2f * v;                    // LeakyReLU(0.2)
  ebuf[(size_t)e * H + hd] = v;
  atomicMax(mmax + (size_t)d * H + hd, f2ord(v));
}

// Pass 2: p = exp(e - m[dst]); stash p; segment sum over dst.
__global__ void edge_exp_sum(const int* __restrict__ eidx, int Eo, int n,
                             const unsigned* __restrict__ mmax, int H,
                             float* __restrict__ ebuf,
                             float* __restrict__ ssum) {
  int t = blockIdx.x * blockDim.x + threadIdx.x;
  int total = (Eo + n) * H;
  if (t >= total) return;
  int e = t / H, hd = t - e * H;
  int d = (e < Eo) ? eidx[Eo + e] : (e - Eo);
  float m = ord2f(mmax[(size_t)d * H + hd]);
  float p = expf(ebuf[(size_t)e * H + hd] - m);
  ebuf[(size_t)e * H + hd] = p;
  atomicAdd(ssum + (size_t)d * H + hd, p);
}

// Pass 3: alpha = p/(s+eps); out[dst,h,:] += alpha * hfeat[src,h,:].
// One wave per (edge, head); lanes cover D=64 contiguously.
__global__ void edge_aggregate(const int* __restrict__ eidx, int Eo, int n,
                               const float* __restrict__ ebuf,
                               const float* __restrict__ ssum,
                               const float* __restrict__ hfeat,
                               float* __restrict__ outagg, int H, int D) {
  int wid  = blockIdx.x * (blockDim.x >> 5) + (threadIdx.x >> 5);
  int lane = threadIdx.x & 31;
  int total = (Eo + n) * H;
  if (wid >= total) return;
  int e = wid / H, hd = wid - e * H;
  int s = (e < Eo) ? eidx[e]      : (e - Eo);
  int d = (e < Eo) ? eidx[Eo + e] : (e - Eo);
  float alpha = ebuf[(size_t)e * H + hd] /
                (ssum[(size_t)d * H + hd] + 1e-16f);
  const float* hs = hfeat + ((size_t)s * H + hd) * D;
  float*       od = outagg + ((size_t)d * H + hd) * D;
  for (int j = lane; j < D; j += 32)
    atomicAdd(od + j, alpha * hs[j]);
}

// out[i] = relu(out[i] + bias[i % C]), in place.
__global__ void bias_relu(float* __restrict__ x, const float* __restrict__ b,
                          int total, int C) {
  int i = blockIdx.x * blockDim.x + threadIdx.x;
  if (i >= total) return;
  float v = x[i] + b[i % C];
  x[i] = v > 0.f ? v : 0.f;
}

// Global mean pool, phase 1: per-node scatter-add of features + counts.
__global__ void pool_sum(const float* __restrict__ h, const int* __restrict__ batch,
                         float* __restrict__ psum, float* __restrict__ pcnt,
                         int n, int D) {
  int wid  = blockIdx.x * (blockDim.x >> 5) + (threadIdx.x >> 5);
  int lane = threadIdx.x & 31;
  if (wid >= n) return;
  int g = batch[wid];
  const float* hp = h + (size_t)wid * D;
  float* ps = psum + (size_t)g * D;
  for (int j = lane; j < D; j += 32) atomicAdd(ps + j, hp[j]);
  if (lane == 0) atomicAdd(pcnt + g, 1.0f);
}

// Pool finalize + classifier + log_softmax. One thread per graph.
__global__ void classify(const float* __restrict__ psum,
                         const float* __restrict__ pcnt,
                         const float* __restrict__ Wc,
                         const float* __restrict__ bc,
                         float* __restrict__ out, int ng, int D) {
  int g = blockIdx.x * blockDim.x + threadIdx.x;
  if (g >= ng) return;
  float inv = 1.0f / fmaxf(pcnt[g], 1.0f);
  float l0 = bc[0], l1 = bc[1];
  const float* ps = psum + (size_t)g * D;
  for (int dd = 0; dd < D; ++dd) {
    float pv = ps[dd] * inv;
    l0 += pv * Wc[dd * 2 + 0];
    l1 += pv * Wc[dd * 2 + 1];
  }
  float mx = fmaxf(l0, l1);
  float lse = mx + logf(expf(l0 - mx) + expf(l1 - mx));
  out[(size_t)g * 2 + 0] = l0 - lse;
  out[(size_t)g * 2 + 1] = l1 - lse;
}

// ---------------------------------------------------------------------------
static inline int cdiv(long long a, long long b) { return (int)((a + b - 1) / b); }

extern "C" void kernel_launch(void* const* d_in, const int* in_sizes, int n_in,
                              void* d_out, int out_size, void* d_ws, size_t ws_size,
                              hipStream_t stream) {
  const float* x      = (const float*)d_in[0];
  const int*   eidx   = (const int*)d_in[1];     // [2,E] flat: [0..E)=src, [E..2E)=dst
  const int*   batch  = (const int*)d_in[2];
  const float* W1     = (const float*)d_in[3];   // [128,256]
  const float* a_src1 = (const float*)d_in[4];   // [4,64]
  const float* a_dst1 = (const float*)d_in[5];
  const float* b1     = (const float*)d_in[6];   // [256]
  const float* W2     = (const float*)d_in[7];   // [256,64]
  const float* a_src2 = (const float*)d_in[8];   // [1,64]
  const float* a_dst2 = (const float*)d_in[9];
  const float* b2     = (const float*)d_in[10];  // [64]
  const float* Wc     = (const float*)d_in[11];  // [64,2]
  const float* bc     = (const float*)d_in[12];  // [2]
  float* out = (float*)d_out;

  const int n = N_NODES, E = NEDGE, E2 = NEDGE + N_NODES;

  // ---- workspace bump allocator ----
  size_t off = 0;
  auto alloc = [&](size_t bytes) -> void* {
    void* p = (char*)d_ws + off;
    off += (bytes + 255) & ~(size_t)255;
    return p;
  };
  float*    h1    = (float*)alloc((size_t)n * 256 * 4);   // layer1 GEMM out
  float*    agg1  = (float*)alloc((size_t)n * 256 * 4);   // layer1 aggregate
  float*    asrc  = (float*)alloc((size_t)n * HEADS1 * 4);
  float*    adst  = (float*)alloc((size_t)n * HEADS1 * 4);
  unsigned* mmax  = (unsigned*)alloc((size_t)n * HEADS1 * 4);
  float*    ssum  = (float*)alloc((size_t)n * HEADS1 * 4);
  float*    ebuf  = (float*)alloc((size_t)E2 * HEADS1 * 4);
  float*    psum  = (float*)alloc((size_t)NGRAPH * HID * 4);
  float*    pcnt  = (float*)alloc((size_t)NGRAPH * 4);
  // layer2 reuses the (now free) h1 region: h2 [N,64] and agg2 [N,64]
  float* h2   = h1;                     // 12.8 MB into 51.2 MB region
  float* agg2 = h1 + (size_t)n * 64;

  const unsigned ORD_NEG_INF = 0x007FFFFFu;  // f2ord(-inf)

  // ================= Layer 1 =================
  {
    dim3 g1(n / 16, 256 / 16);
    gemm_wmma_f16<F_IN, 256><<<g1, 32, 0, stream>>>(x, W1, h1);

    int t = n * HEADS1;
    node_att_kernel<<<cdiv(t, 256), 256, 0, stream>>>(h1, a_src1, a_dst1,
                                                      asrc, adst, n, HEADS1, HID);
    fill_u32<<<cdiv(t, 256), 256, 0, stream>>>(mmax, ORD_NEG_INF, (size_t)t);
    fill_f32<<<cdiv(t, 256), 256, 0, stream>>>(ssum, 0.f, (size_t)t);
    fill_f32<<<cdiv((long long)n * 256, 256), 256, 0, stream>>>(agg1, 0.f, (size_t)n * 256);

    int te = E2 * HEADS1;
    edge_logits_max<<<cdiv(te, 256), 256, 0, stream>>>(eidx, E, n, asrc, adst,
                                                       HEADS1, ebuf, mmax);
    edge_exp_sum<<<cdiv(te, 256), 256, 0, stream>>>(eidx, E, n, mmax, HEADS1,
                                                    ebuf, ssum);
    edge_aggregate<<<cdiv(te, 8), 256, 0, stream>>>(eidx, E, n, ebuf, ssum, h1,
                                                    agg1, HEADS1, HID);
    bias_relu<<<cdiv((long long)n * 256, 256), 256, 0, stream>>>(agg1, b1, n * 256, 256);
  }

  // ================= Layer 2 =================
  {
    dim3 g2(n / 16, 64 / 16);
    gemm_wmma_f16<256, HID><<<g2, 32, 0, stream>>>(agg1, W2, h2);

    node_att_kernel<<<cdiv(n, 256), 256, 0, stream>>>(h2, a_src2, a_dst2,
                                                      asrc, adst, n, 1, HID);
    fill_u32<<<cdiv(n, 256), 256, 0, stream>>>(mmax, ORD_NEG_INF, (size_t)n);
    fill_f32<<<cdiv(n, 256), 256, 0, stream>>>(ssum, 0.f, (size_t)n);
    fill_f32<<<cdiv((long long)n * 64, 256), 256, 0, stream>>>(agg2, 0.f, (size_t)n * 64);

    edge_logits_max<<<cdiv(E2, 256), 256, 0, stream>>>(eidx, E, n, asrc, adst,
                                                       1, ebuf, mmax);
    edge_exp_sum<<<cdiv(E2, 256), 256, 0, stream>>>(eidx, E, n, mmax, 1,
                                                    ebuf, ssum);
    edge_aggregate<<<cdiv(E2, 8), 256, 0, stream>>>(eidx, E, n, ebuf, ssum, h2,
                                                    agg2, 1, HID);
    bias_relu<<<cdiv((long long)n * 64, 256), 256, 0, stream>>>(agg2, b2, n * 64, HID);
  }

  // ================= Pool + classifier =================
  fill_f32<<<cdiv(NGRAPH * HID, 256), 256, 0, stream>>>(psum, 0.f, (size_t)NGRAPH * HID);
  fill_f32<<<cdiv(NGRAPH, 256), 256, 0, stream>>>(pcnt, 0.f, (size_t)NGRAPH);
  pool_sum<<<cdiv(n, 8), 256, 0, stream>>>(agg2, batch, psum, pcnt, n, HID);
  classify<<<cdiv(NGRAPH, 256), 256, 0, stream>>>(psum, pcnt, Wc, bc, out,
                                                  NGRAPH, HID);
}
